// Stage0_12532714569976
// MI455X (gfx1250) — matrix operand order repaired and checked by reading, not compiled, not run on hardware
//
#include <hip/hip_runtime.h>
#include <math.h>

// ---------------- model constants ----------------
#define B_   4
#define L_   2048
#define H_   128
#define I_   256
#define N_   16
#define R_   8
#define K_   4
#define F_   14336
#define M_   (B_ * L_)          // 8192 tokens
#define TWO_I (2 * I_)          // 512
#define SSMW  (R_ + 2 * N_)     // 40
#define EPSV 1e-6f

// ---------------- WMMA types ----------------
typedef __attribute__((ext_vector_type(16))) __bf16        v16bf;
typedef __attribute__((ext_vector_type(8)))  float         v8f;
typedef __attribute__((ext_vector_type(8)))  unsigned int  v8u;

#define WMMA_BF16(a, b, c) \
  __builtin_amdgcn_wmma_f32_16x16x32_bf16(false, (a), false, (b), (short)0, (c), false, false)

// ---------------- helpers ----------------
__device__ __forceinline__ unsigned short f2bf(float f) {
  unsigned int u = __float_as_uint(f);
  u += 0x7FFFu + ((u >> 16) & 1u);          // round-to-nearest-even
  return (unsigned short)(u >> 16);
}
__device__ __forceinline__ float siluf(float x) { return x / (1.f + expf(-x)); }

// ---------------- GEMM tiling ----------------
#define BLK_M 128
#define BLK_N 64
#define TILE_K 32
#define LDS_PITCH 40   // ushorts per LDS row: 64B data + 16B pad (16B-aligned rows)

// A fragment (16x32 bf16, ISA 7.12.2): lane m = lane&15, kBase=(lane>>4)*8,
// elems 0..7 -> K=kBase..kBase+7, elems 8..15 -> K=kBase+16..kBase+23.
__device__ __forceinline__ v16bf load_a_frag(const unsigned short* At, int mloc, int aKBase) {
  const uint4* p0 = (const uint4*)(At + mloc * LDS_PITCH + aKBase);
  const uint4* p1 = (const uint4*)(At + mloc * LDS_PITCH + aKBase + 16);
  uint4 x0 = *p0, x1 = *p1;
  v8u v;
  v[0] = x0.x; v[1] = x0.y; v[2] = x0.z; v[3] = x0.w;
  v[4] = x1.x; v[5] = x1.y; v[6] = x1.z; v[7] = x1.w;
  return __builtin_bit_cast(v16bf, v);
}
// B fragment (32x16 bf16): lane n = lane&15, kBase=(lane>>4)*8, elems 0..7 -> K=kBase..kBase+7, elems 8..15 -> K=kBase+16..kBase+23.
__device__ __forceinline__ v16bf load_b_frag(const unsigned short* Wt, int nloc, int bKBase) {
  const uint4* p0 = (const uint4*)(Wt + nloc * LDS_PITCH + bKBase);
  const uint4* p1 = (const uint4*)(Wt + nloc * LDS_PITCH + bKBase + 16);
  uint4 x0 = *p0, x1 = *p1;
  v8u v;
  v[0] = x0.x; v[1] = x0.y; v[2] = x0.z; v[3] = x0.w;
  v[4] = x1.x; v[5] = x1.y; v[6] = x1.z; v[7] = x1.w;
  return __builtin_bit_cast(v16bf, v);
}

// ---------------- Tensor Data Mover staging (CDNA5 TDM, ISA ch.8) ----------------
#if defined(__HIP_DEVICE_COMPILE__) && __has_builtin(__builtin_amdgcn_tensor_load_to_lds)
#define HAVE_TDM 1
typedef __attribute__((ext_vector_type(4))) unsigned int u32x4;
typedef __attribute__((ext_vector_type(8))) int          i32x8;
typedef __attribute__((ext_vector_type(4))) int          i32x4;

// 2D tile load: tile_x (elems, fastest dim) x tile_y rows of 2-byte elements,
// row stride = row_stride elems, remaining valid extent (rem_x, rem_y) for OOB zero-fill,
// LDS padded 16B per 64B row (pad_interval=16 dwords -> code 3, pad_amount=4 dwords -> code 3).
__device__ __forceinline__ void tdm_load_tile_2d(unsigned int lds_addr,
                                                 const unsigned short* gptr,
                                                 unsigned int tile_x, unsigned int tile_y,
                                                 unsigned int rem_x, unsigned int rem_y,
                                                 unsigned long long row_stride) {
  unsigned long long ga = (unsigned long long)(uintptr_t)gptr;
  u32x4 g0;
  g0[0] = 1u;                                                    // count=1 (valid)
  g0[1] = lds_addr;                                              // lds_addr[31:0]
  g0[2] = (unsigned int)ga;                                      // global_addr[31:0]
  g0[3] = (unsigned int)((ga >> 32) & 0x01FFFFFFu) | (2u << 30); // addr[56:32] | type=2
  unsigned int d[8];
  d[0] = (1u << 16)            // data_size = 1 -> 2 bytes
       | (1u << 20)            // pad_enable
       | (3u << 22)            // pad_interval code: 16 dwords (64B)
       | (3u << 25);           // pad_amount code: 4 dwords (16B)
  d[1] = (rem_x & 0xFFFFu) << 16;                      // tensor_dim0[15:0] at bits[63:48]
  d[2] = (rem_x >> 16) | ((rem_y & 0xFFFFu) << 16);    // dim0 hi | dim1 lo
  d[3] = (rem_y >> 16) | (tile_x << 16);               // dim1 hi | tile_dim0
  d[4] = tile_y & 0xFFFFu;                             // tile_dim1 (tile_dim2 = 0)
  d[5] = (unsigned int)row_stride;                     // tensor_dim0_stride[31:0]
  d[6] = (unsigned int)(row_stride >> 32) & 0xFFFFu;   // stride[47:32]
  d[7] = 0u;
  i32x8 g1;
  for (int q = 0; q < 8; ++q) g1[q] = (int)d[q];
  i32x4 g2 = {0, 0, 0, 0}, g3 = {0, 0, 0, 0};
#if __clang_major__ >= 23
  i32x8 g4 = {0, 0, 0, 0, 0, 0, 0, 0};
  __builtin_amdgcn_tensor_load_to_lds(g0, g1, g2, g3, g4, 0);
#else
  __builtin_amdgcn_tensor_load_to_lds(g0, g1, g2, g3, 0);
#endif
}
__device__ __forceinline__ unsigned int lds_addr_of(const void* p) {
  return (unsigned int)(uintptr_t)p;   // generic ptr low 32 bits = LDS offset
}
#endif

// manual staging fallbacks (global_load_b128 -> ds_store_b128)
__device__ __forceinline__ void stage_manual_A(unsigned short* dst, const unsigned short* A,
                                               size_t rowBase, int Kd, int k0, int tid, int rows) {
  for (int t = tid; t < rows * 4; t += 256) {
    int row = t >> 2, c8 = t & 3;
    uint4 v = *(const uint4*)(A + (rowBase + row) * (size_t)Kd + k0 + c8 * 8);
    *(uint4*)(dst + row * LDS_PITCH + c8 * 8) = v;
  }
}
__device__ __forceinline__ void stage_manual_W(unsigned short* dst, const unsigned short* W,
                                               int nBase, int Nd, int Kd, int k0, int tid, int rows) {
  for (int t = tid; t < rows * 4; t += 256) {
    int row = t >> 2, c8 = t & 3;
    int n = nBase + row;
    uint4 v = make_uint4(0u, 0u, 0u, 0u);
    if (n < Nd) v = *(const uint4*)(W + (size_t)n * Kd + k0 + c8 * 8);
    *(uint4*)(dst + row * LDS_PITCH + c8 * 8) = v;
  }
}

// ---------------- generic bf16 GEMM: C[M,N] = A[M,K] @ W[N,K]^T (+ addend) ----------------
// Block tile 128x64, 8 waves (4x2), wave tile 32x32 (4 WMMAs / K-step), LDS double-buffered.
// Requires: M % 128 == 0, K % 32 == 0. N bounds-checked (TDM OOB zero-fill / guarded loads).
__global__ __launch_bounds__(256) void gemm_bf16_kernel(
    const unsigned short* __restrict__ A, const unsigned short* __restrict__ W,
    const float* __restrict__ addend, float* __restrict__ C, int M, int N, int Kd) {
  __shared__ __align__(16) unsigned short At[2][BLK_M * LDS_PITCH];
  __shared__ __align__(16) unsigned short Wt[2][BLK_N * LDS_PITCH];
  const int tid  = threadIdx.x;
  const int lane = tid & 31;
  const int wave = tid >> 5;
  const int wm = wave >> 1, wn = wave & 1;        // 4(M) x 2(N) waves
  const int mBase = blockIdx.y * BLK_M;
  const int nBase = blockIdx.x * BLK_N;

  v8f acc00 = {}, acc01 = {}, acc10 = {}, acc11 = {};
  const int aKBase = (lane >> 4) * 8;
  const int bKBase = (lane >> 4) * 8;
  const int mloc0  = wm * 32 + (lane & 15);
  const int nloc0  = wn * 32 + (lane & 15);

  // stage(buf, k0): bring A/W K-slice into LDS buffer
#ifdef HAVE_TDM
  #define STAGE(buf, kk)                                                              \
    do {                                                                              \
      if (wave == 0)                                                                  \
        tdm_load_tile_2d(lds_addr_of(&At[buf][0]), A + (size_t)mBase * Kd + (kk),     \
                         TILE_K, BLK_M, (unsigned)(Kd - (kk)), (unsigned)(M - mBase), \
                         (unsigned long long)Kd);                                     \
      else if (wave == 1)                                                             \
        tdm_load_tile_2d(lds_addr_of(&Wt[buf][0]), W + (size_t)nBase * Kd + (kk),     \
                         TILE_K, BLK_N, (unsigned)(Kd - (kk)),                        \
                         (unsigned)(N - nBase), (unsigned long long)Kd);              \
    } while (0)
  #define STAGE_WAIT() do { if (wave <= 1) __builtin_amdgcn_s_wait_tensorcnt(0); } while (0)
#else
  #define STAGE(buf, kk)                                                   \
    do {                                                                   \
      stage_manual_A(&At[buf][0], A, (size_t)mBase, Kd, (kk), tid, BLK_M); \
      stage_manual_W(&Wt[buf][0], W, nBase, N, Kd, (kk), tid, BLK_N);      \
    } while (0)
  #define STAGE_WAIT() do { } while (0)
#endif

  int buf = 0;
  STAGE(0, 0);
  STAGE_WAIT();
  __syncthreads();
  for (int k0 = 0; k0 < Kd; k0 += TILE_K) {
    int nk = k0 + TILE_K;
    if (nk < Kd) STAGE(buf ^ 1, nk);          // async prefetch of next K-slice
    const unsigned short* Ab = &At[buf][0];
    const unsigned short* Wb = &Wt[buf][0];
    v16bf a0 = load_a_frag(Ab, mloc0, aKBase);
    v16bf a1 = load_a_frag(Ab, mloc0 + 16, aKBase);
    v16bf b0 = load_b_frag(Wb, nloc0, bKBase);
    v16bf b1 = load_b_frag(Wb, nloc0 + 16, bKBase);
    acc00 = WMMA_BF16(a0, b0, acc00);
    acc01 = WMMA_BF16(a0, b1, acc01);
    acc10 = WMMA_BF16(a1, b0, acc10);
    acc11 = WMMA_BF16(a1, b1, acc11);
    STAGE_WAIT();
    __syncthreads();
    buf ^= 1;
  }
#undef STAGE
#undef STAGE_WAIT

  const int rbase = (lane >> 4) * 8;
  v8f* accs[2][2] = {{&acc00, &acc01}, {&acc10, &acc11}};
  for (int i = 0; i < 2; ++i)
    for (int j = 0; j < 2; ++j) {
      int n = nBase + wn * 32 + j * 16 + (lane & 15);
      if (n >= N) continue;
      int mRow = mBase + wm * 32 + i * 16 + rbase;
      if (addend) {
        for (int r = 0; r < 8; ++r) {
          size_t idx = (size_t)(mRow + r) * N + n;
          C[idx] = (*accs[i][j])[r] + addend[idx];
        }
      } else {
        for (int r = 0; r < 8; ++r) C[(size_t)(mRow + r) * N + n] = (*accs[i][j])[r];
      }
    }
}

// ---------------- fused FFN gate+up GEMM -> act = silu(h@Wg^T) * (h@Wu^T) as bf16 ----------------
__global__ __launch_bounds__(256) void gemm_gateup_kernel(
    const unsigned short* __restrict__ A, const unsigned short* __restrict__ Wg,
    const unsigned short* __restrict__ Wu, unsigned short* __restrict__ act,
    int M, int N, int Kd) {
  __shared__ __align__(16) unsigned short At [2][BLK_M * LDS_PITCH];
  __shared__ __align__(16) unsigned short Wgt[2][BLK_N * LDS_PITCH];
  __shared__ __align__(16) unsigned short Wut[2][BLK_N * LDS_PITCH];
  const int tid  = threadIdx.x;
  const int lane = tid & 31;
  const int wave = tid >> 5;
  const int wm = wave >> 1, wn = wave & 1;
  const int mBase = blockIdx.y * BLK_M;
  const int nBase = blockIdx.x * BLK_N;

  v8f g00 = {}, g01 = {}, g10 = {}, g11 = {};
  v8f u00 = {}, u01 = {}, u10 = {}, u11 = {};
  const int aKBase = (lane >> 4) * 8;
  const int bKBase = (lane >> 4) * 8;
  const int mloc0  = wm * 32 + (lane & 15);
  const int nloc0  = wn * 32 + (lane & 15);

#ifdef HAVE_TDM
  #define STAGE(buf, kk)                                                               \
    do {                                                                               \
      if (wave == 0)                                                                   \
        tdm_load_tile_2d(lds_addr_of(&At[buf][0]), A + (size_t)mBase * Kd + (kk),      \
                         TILE_K, BLK_M, (unsigned)(Kd - (kk)), (unsigned)(M - mBase),  \
                         (unsigned long long)Kd);                                      \
      else if (wave == 1)                                                              \
        tdm_load_tile_2d(lds_addr_of(&Wgt[buf][0]), Wg + (size_t)nBase * Kd + (kk),    \
                         TILE_K, BLK_N, (unsigned)(Kd - (kk)), (unsigned)(N - nBase),  \
                         (unsigned long long)Kd);                                      \
      else if (wave == 2)                                                              \
        tdm_load_tile_2d(lds_addr_of(&Wut[buf][0]), Wu + (size_t)nBase * Kd + (kk),    \
                         TILE_K, BLK_N, (unsigned)(Kd - (kk)), (unsigned)(N - nBase),  \
                         (unsigned long long)Kd);                                      \
    } while (0)
  #define STAGE_WAIT() do { if (wave <= 2) __builtin_amdgcn_s_wait_tensorcnt(0); } while (0)
#else
  #define STAGE(buf, kk)                                                    \
    do {                                                                    \
      stage_manual_A(&At[buf][0], A, (size_t)mBase, Kd, (kk), tid, BLK_M);  \
      stage_manual_W(&Wgt[buf][0], Wg, nBase, N, Kd, (kk), tid, BLK_N);     \
      stage_manual_W(&Wut[buf][0], Wu, nBase, N, Kd, (kk), tid, BLK_N);     \
    } while (0)
  #define STAGE_WAIT() do { } while (0)
#endif

  int buf = 0;
  STAGE(0, 0);
  STAGE_WAIT();
  __syncthreads();
  for (int k0 = 0; k0 < Kd; k0 += TILE_K) {
    int nk = k0 + TILE_K;
    if (nk < Kd) STAGE(buf ^ 1, nk);
    const unsigned short* Ab  = &At[buf][0];
    const unsigned short* Wgb = &Wgt[buf][0];
    const unsigned short* Wub = &Wut[buf][0];
    v16bf a0 = load_a_frag(Ab, mloc0, aKBase);
    v16bf a1 = load_a_frag(Ab, mloc0 + 16, aKBase);
    v16bf bg0 = load_b_frag(Wgb, nloc0, bKBase);
    v16bf bg1 = load_b_frag(Wgb, nloc0 + 16, bKBase);
    v16bf bu0 = load_b_frag(Wub, nloc0, bKBase);
    v16bf bu1 = load_b_frag(Wub, nloc0 + 16, bKBase);
    g00 = WMMA_BF16(a0, bg0, g00);
    g01 = WMMA_BF16(a0, bg1, g01);
    g10 = WMMA_BF16(a1, bg0, g10);
    g11 = WMMA_BF16(a1, bg1, g11);
    u00 = WMMA_BF16(a0, bu0, u00);
    u01 = WMMA_BF16(a0, bu1, u01);
    u10 = WMMA_BF16(a1, bu0, u10);
    u11 = WMMA_BF16(a1, bu1, u11);
    STAGE_WAIT();
    __syncthreads();
    buf ^= 1;
  }
#undef STAGE
#undef STAGE_WAIT

  const int rbase = (lane >> 4) * 8;
  v8f* gs[2][2] = {{&g00, &g01}, {&g10, &g11}};
  v8f* us[2][2] = {{&u00, &u01}, {&u10, &u11}};
  for (int i = 0; i < 2; ++i)
    for (int j = 0; j < 2; ++j) {
      int n = nBase + wn * 32 + j * 16 + (lane & 15);
      int mRow = mBase + wm * 32 + i * 16 + rbase;
      for (int r = 0; r < 8; ++r) {
        float g = (*gs[i][j])[r], u = (*us[i][j])[r];
        act[(size_t)(mRow + r) * N + n] = f2bf(siluf(g) * u);
      }
    }
}

// ---------------- f32 -> bf16 weight conversion ----------------
__global__ void f2bf_kernel(const float* __restrict__ src, unsigned short* __restrict__ dst, int n) {
  int i = blockIdx.x * blockDim.x + threadIdx.x;
  if (i < n) dst[i] = f2bf(src[i]);
}
__global__ void aneg_kernel(const float* __restrict__ A_log, float* __restrict__ A_neg, int n) {
  int i = blockIdx.x * blockDim.x + threadIdx.x;
  if (i < n) A_neg[i] = -expf(A_log[i]);
}

// ---------------- embed gather + RMSNorm(ln1) ----------------
__global__ void embed_rms_kernel(const int* __restrict__ ids, const float* __restrict__ embed,
                                 const float* __restrict__ w, float* __restrict__ res,
                                 unsigned short* __restrict__ hbf) {
  int token = blockIdx.x, t = threadIdx.x;         // 128 threads
  int id = ids[token];
  float x = embed[(size_t)id * H_ + t];
  res[(size_t)token * H_ + t] = x;
  __shared__ float sred[H_];
  sred[t] = x * x; __syncthreads();
  for (int s = H_ / 2; s > 0; s >>= 1) { if (t < s) sred[t] += sred[t + s]; __syncthreads(); }
  float inv = rsqrtf(sred[0] / (float)H_ + EPSV);
  hbf[(size_t)token * H_ + t] = f2bf(x * inv * w[t]);
}

// ---------------- RMSNorm(ln2) from f32 buffer ----------------
__global__ void rms_kernel(const float* __restrict__ xin, const float* __restrict__ w,
                           unsigned short* __restrict__ hbf) {
  int token = blockIdx.x, t = threadIdx.x;         // 128 threads
  float x = xin[(size_t)token * H_ + t];
  __shared__ float sred[H_];
  sred[t] = x * x; __syncthreads();
  for (int s = H_ / 2; s > 0; s >>= 1) { if (t < s) sred[t] += sred[t + s]; __syncthreads(); }
  float inv = rsqrtf(sred[0] / (float)H_ + EPSV);
  hbf[(size_t)token * H_ + t] = f2bf(x * inv * w[t]);
}

// ---------------- depthwise causal conv (K=4) + bias + SiLU ----------------
__global__ void conv_silu_kernel(const float* __restrict__ proj, const float* __restrict__ conv_w,
                                 const float* __restrict__ conv_b, float* __restrict__ hs_f,
                                 unsigned short* __restrict__ hs_b) {
  int token = blockIdx.x, i = threadIdx.x;         // 256 threads
  int b = token / L_, l = token % L_;
  float acc = conv_b[i];
  #pragma unroll
  for (int k = 0; k < K_; ++k) {
    int lk = l - (K_ - 1) + k;
    if (lk >= 0)
      acc += conv_w[i * K_ + k] * proj[(size_t)(b * L_ + lk) * TWO_I + i];
  }
  float s = siluf(acc);
  hs_f[(size_t)token * I_ + i] = s;
  hs_b[(size_t)token * I_ + i] = f2bf(s);
}

// ---------------- tiny RMSNorms (dt/B/C) + dt_proj + softplus ----------------
__global__ void ssm_prep_kernel(const float* __restrict__ ssm, const float* __restrict__ dt_ln_w,
                                const float* __restrict__ b_ln_w, const float* __restrict__ c_ln_w,
                                const float* __restrict__ dt_proj_w, const float* __restrict__ dt_proj_b,
                                float* __restrict__ bm, float* __restrict__ cm,
                                float* __restrict__ dtf) {
  int token = blockIdx.x, t = threadIdx.x;         // 256 threads
  __shared__ float sv[SSMW];
  __shared__ float dtn[R_];
  __shared__ float inv3[3];
  if (t < SSMW) sv[t] = ssm[(size_t)token * SSMW + t];
  __syncthreads();
  if (t == 0) { float s = 0; for (int r = 0; r < R_; ++r) s += sv[r] * sv[r];
                inv3[0] = rsqrtf(s / (float)R_ + EPSV); }
  if (t == 1) { float s = 0; for (int r = 0; r < N_; ++r) s += sv[R_ + r] * sv[R_ + r];
                inv3[1] = rsqrtf(s / (float)N_ + EPSV); }
  if (t == 2) { float s = 0; for (int r = 0; r < N_; ++r) s += sv[R_ + N_ + r] * sv[R_ + N_ + r];
                inv3[2] = rsqrtf(s / (float)N_ + EPSV); }
  __syncthreads();
  if (t < R_)                     dtn[t] = sv[t] * inv3[0] * dt_ln_w[t];
  else if (t < R_ + N_)           bm[(size_t)token * N_ + (t - R_)]      = sv[t] * inv3[1] * b_ln_w[t - R_];
  else if (t < SSMW)              cm[(size_t)token * N_ + (t - R_ - N_)] = sv[t] * inv3[2] * c_ln_w[t - R_ - N_];
  __syncthreads();
  float acc = dt_proj_b[t];
  #pragma unroll
  for (int r = 0; r < R_; ++r) acc += dtn[r] * dt_proj_w[t * R_ + r];
  dtf[(size_t)token * I_ + t] = (acc > 20.f) ? acc : log1pf(expf(acc));
}

// ---------------- selective scan + D skip + SiLU gating -> bf16 ----------------
__global__ void scan_kernel(const float* __restrict__ dtf, const float* __restrict__ bm,
                            const float* __restrict__ cm, const float* __restrict__ hs_f,
                            const float* __restrict__ proj, const float* __restrict__ A_neg,
                            const float* __restrict__ D, unsigned short* __restrict__ y_b) {
  int gid = blockIdx.x * blockDim.x + threadIdx.x;
  int pair = gid >> 4;                 // (b,i) index: 0..B_*I_-1
  int nlane = threadIdx.x & 15;
  if (pair >= B_ * I_) return;
  int b = pair >> 8;                   // / I_
  int i = pair & (I_ - 1);
  float a = A_neg[i * N_ + nlane];
  float d = D[i];
  float s = 0.f;
  size_t tokBase = (size_t)b * L_;
  for (int l = 0; l < L_; ++l) {
    size_t tok = tokBase + l;
    float dt  = dtf[tok * I_ + i];
    float hsv = hs_f[tok * I_ + i];
    float bmv = bm[tok * N_ + nlane];
    float cmv = cm[tok * N_ + nlane];
    float dA = expf(dt * a);
    s = dA * s + dt * bmv * hsv;
    float yp = s * cmv;
    yp += __shfl_xor(yp, 1, 32);
    yp += __shfl_xor(yp, 2, 32);
    yp += __shfl_xor(yp, 4, 32);
    yp += __shfl_xor(yp, 8, 32);
    if (nlane == 0) {
      float g = proj[tok * TWO_I + I_ + i];
      float y = (yp + hsv * d) * siluf(g);
      y_b[tok * I_ + i] = f2bf(y);
    }
  }
}

// ---------------- launch ----------------
static inline size_t alignup(size_t x) { return (x + 255) & ~(size_t)255; }

extern "C" void kernel_launch(void* const* d_in, const int* in_sizes, int n_in,
                              void* d_out, int out_size, void* d_ws, size_t ws_size,
                              hipStream_t stream) {
  const int*   input_ids = (const int*)  d_in[0];
  const float* embed     = (const float*)d_in[1];
  const float* ln1_w     = (const float*)d_in[2];
  const float* in_proj_w = (const float*)d_in[3];
  const float* conv_w    = (const float*)d_in[4];
  const float* conv_b    = (const float*)d_in[5];
  const float* x_proj_w  = (const float*)d_in[6];
  const float* dt_ln_w   = (const float*)d_in[7];
  const float* b_ln_w    = (const float*)d_in[8];
  const float* c_ln_w    = (const float*)d_in[9];
  const float* dt_proj_w = (const float*)d_in[10];
  const float* dt_proj_b = (const float*)d_in[11];
  const float* A_log     = (const float*)d_in[12];
  const float* Dp        = (const float*)d_in[13];
  const float* out_proj_w= (const float*)d_in[14];
  const float* ln2_w     = (const float*)d_in[15];
  const float* gate_w    = (const float*)d_in[16];
  const float* up_w      = (const float*)d_in[17];
  const float* down_w    = (const float*)d_in[18];
  float* out = (float*)d_out;

  // ---- workspace carve ----
  char* ws = (char*)d_ws;  size_t off = 0;
  auto carve = [&](size_t bytes) { char* p = ws + off; off = alignup(off + bytes); return p; };
  float*          res1 = (float*)         carve((size_t)M_ * H_ * 4);
  unsigned short* h1   = (unsigned short*)carve((size_t)M_ * H_ * 2);
  float*          proj = (float*)         carve((size_t)M_ * TWO_I * 4);
  float*          hs_f = (float*)         carve((size_t)M_ * I_ * 4);
  unsigned short* hs_b = (unsigned short*)carve((size_t)M_ * I_ * 2);
  float*          ssm  = (float*)         carve((size_t)M_ * SSMW * 4);
  float*          bm   = (float*)         carve((size_t)M_ * N_ * 4);
  float*          cm   = (float*)         carve((size_t)M_ * N_ * 4);
  float*          dtf  = (float*)         carve((size_t)M_ * I_ * 4);
  unsigned short* y_b  = (unsigned short*)carve((size_t)M_ * I_ * 2);
  float*          x2   = (float*)         carve((size_t)M_ * H_ * 4);
  unsigned short* h2   = (unsigned short*)carve((size_t)M_ * H_ * 2);
  unsigned short* act  = (unsigned short*)carve((size_t)M_ * F_ * 2);
  unsigned short* w_in = (unsigned short*)carve((size_t)TWO_I * H_ * 2);
  unsigned short* w_xp = (unsigned short*)carve((size_t)SSMW * I_ * 2);
  unsigned short* w_op = (unsigned short*)carve((size_t)H_ * I_ * 2);
  unsigned short* w_g  = (unsigned short*)carve((size_t)F_ * H_ * 2);
  unsigned short* w_u  = (unsigned short*)carve((size_t)F_ * H_ * 2);
  unsigned short* w_d  = (unsigned short*)carve((size_t)H_ * F_ * 2);
  float*          A_neg= (float*)         carve((size_t)I_ * N_ * 4);
  (void)ws_size; (void)n_in; (void)in_sizes; (void)out_size;

  // ---- weight conversions (f32 -> bf16) ----
  auto conv1d = [&](const float* s, unsigned short* dst, int n) {
    f2bf_kernel<<<(n + 255) / 256, 256, 0, stream>>>(s, dst, n);
  };
  conv1d(in_proj_w,  w_in, TWO_I * H_);
  conv1d(x_proj_w,   w_xp, SSMW * I_);
  conv1d(out_proj_w, w_op, H_ * I_);
  conv1d(gate_w,     w_g,  F_ * H_);
  conv1d(up_w,       w_u,  F_ * H_);
  conv1d(down_w,     w_d,  H_ * F_);
  aneg_kernel<<<(I_ * N_ + 255) / 256, 256, 0, stream>>>(A_log, A_neg, I_ * N_);

  // ---- 1. embed gather + RMSNorm(ln1) ----
  embed_rms_kernel<<<M_, H_, 0, stream>>>(input_ids, embed, ln1_w, res1, h1);

  // ---- 2. in_proj: proj[M,512] = h1 @ in_proj_w^T ----
  gemm_bf16_kernel<<<dim3(TWO_I / BLK_N, M_ / BLK_M), 256, 0, stream>>>(
      h1, w_in, nullptr, proj, M_, TWO_I, H_);

  // ---- 3. depthwise causal conv + SiLU ----
  conv_silu_kernel<<<M_, I_, 0, stream>>>(proj, conv_w, conv_b, hs_f, hs_b);

  // ---- 4. x_proj: ssm[M,40] = hs @ x_proj_w^T ----
  gemm_bf16_kernel<<<dim3((SSMW + BLK_N - 1) / BLK_N, M_ / BLK_M), 256, 0, stream>>>(
      hs_b, w_xp, nullptr, ssm, M_, SSMW, I_);

  // ---- 5. tiny norms + dt_proj + softplus ----
  ssm_prep_kernel<<<M_, I_, 0, stream>>>(ssm, dt_ln_w, b_ln_w, c_ln_w,
                                         dt_proj_w, dt_proj_b, bm, cm, dtf);

  // ---- 6. selective scan + gating -> y_b (bf16) ----
  scan_kernel<<<(B_ * I_ * 16 + 255) / 256, 256, 0, stream>>>(
      dtf, bm, cm, hs_f, proj, A_neg, Dp, y_b);

  // ---- 7. out_proj + residual: x2 = res1 + y @ out_proj_w^T ----
  gemm_bf16_kernel<<<dim3(H_ / BLK_N, M_ / BLK_M), 256, 0, stream>>>(
      y_b, w_op, res1, x2, M_, H_, I_);

  // ---- 8. RMSNorm(ln2) ----
  rms_kernel<<<M_, H_, 0, stream>>>(x2, ln2_w, h2);

  // ---- 9. fused FFN gate+up: act = silu(h2@Wg^T) * (h2@Wu^T) (bf16) ----
  gemm_gateup_kernel<<<dim3(F_ / BLK_N, M_ / BLK_M), 256, 0, stream>>>(
      h2, w_g, w_u, act, M_, F_, H_);

  // ---- 10. down proj + residual: out = x2 + act @ down_w^T ----
  gemm_bf16_kernel<<<dim3(H_ / BLK_N, M_ / BLK_M), 256, 0, stream>>>(
      act, w_d, x2, out, M_, H_, F_);
}
